// GCN_61838939127865
// MI455X (gfx1250) — matrix operand order repaired and checked by reading, not compiled, not run on hardware
//
#include <hip/hip_runtime.h>
#include <hip/hip_bf16.h>

// ---------------------------------------------------------------------------
// GCN on MI455X (gfx1250). GEMMs run on V_WMMA_F32_16X16X32_F16 (f16 in,
// f32 accumulate); graph scatter runs as L2-resident float atomics.
// ---------------------------------------------------------------------------

#define N_NODES 10000
#define E_EDGES 160000

typedef __attribute__((ext_vector_type(16))) _Float16 v16h;
typedef __attribute__((ext_vector_type(8)))  float    v8f;

// ---------------------------------------------------------------------------
// WMMA GEMM:  C[N x dout] = A[N x K] (f16, row-major) * Wt[dout x K]^T (f16)
// Wt is W pre-transposed so each lane's B-fragment is a contiguous 16-half run.
// Block = 128 threads = 4 waves; each wave owns TWO 16x16 tiles (cols n, n+64)
// sharing one A fragment -> 2 independent WMMAs per K-step, half the A loads.
// Block covers a 16 x 128 slab. grid = (N/16, dout/128).
// ---------------------------------------------------------------------------
__global__ __launch_bounds__(128)
void gcn_wmma_gemm(const _Float16* __restrict__ A,
                   const _Float16* __restrict__ Wt,
                   float* __restrict__ C, int K, int dout) {
    const int wave   = threadIdx.x >> 5;   // 0..3
    const int lane   = threadIdx.x & 31;
    const int m      = lane & 15;          // row within tile (A) / col within tile (B)
    const int hiHalf = lane >> 4;          // lanes 16..31 hold the upper K sub-block
    const int mBase  = blockIdx.x * 16;
    const int n0     = blockIdx.y * 128 + wave * 16;
    const int n1     = n0 + 64;

    const _Float16* __restrict__ Arow = A  + (size_t)(mBase + m) * K;
    const _Float16* __restrict__ B0r  = Wt + (size_t)(n0 + m) * K;
    const _Float16* __restrict__ B1r  = Wt + (size_t)(n1 + m) * K;

    v8f acc0 = {};
    v8f acc1 = {};
    for (int kb = 0; kb < K; kb += 32) {
        // A fragment: lanes 0-15 -> K = kb+0..15 ; lanes 16-31 -> K = kb+16..31
        const int ka = kb + hiHalf * 16;
        union { v16h v; float4 f4[2]; } ua;
        ua.f4[0] = *reinterpret_cast<const float4*>(Arow + ka);
        ua.f4[1] = *reinterpret_cast<const float4*>(Arow + ka + 8);

        // B fragments: lanes 0-15 -> K = kb+0..15 ; lanes 16-31 -> K = kb+16..31
        const int kbb = kb + hiHalf * 16;
        union { v16h v; float4 f4[2]; } ub0, ub1;
        ub0.f4[0] = *reinterpret_cast<const float4*>(B0r + kbb);
        ub0.f4[1] = *reinterpret_cast<const float4*>(B0r + kbb + 8);
        ub1.f4[0] = *reinterpret_cast<const float4*>(B1r + kbb);
        ub1.f4[1] = *reinterpret_cast<const float4*>(B1r + kbb + 8);

        // hint the next K-block into cache (global_prefetch_b8)
        if (kb + 32 < K) {
            __builtin_prefetch(Arow + ka + 32, 0, 3);
            __builtin_prefetch(B0r + kbb + 32, 0, 3);
            __builtin_prefetch(B1r + kbb + 32, 0, 3);
        }

        // two independent WMMAs (no D->A/B hazard between them)
        acc0 = __builtin_amdgcn_wmma_f32_16x16x32_f16(
            false, ua.v, false, ub0.v, (short)0, acc0, false, false);
        acc1 = __builtin_amdgcn_wmma_f32_16x16x32_f16(
            false, ua.v, false, ub1.v, (short)0, acc1, false, false);
    }

    // C/D layout: acc[r] is row (r + 8*hiHalf), col (lane & 15) of the tile
    float* Cb0 = C + (size_t)mBase * dout + n0 + (lane & 15);
    float* Cb1 = C + (size_t)mBase * dout + n1 + (lane & 15);
    #pragma unroll
    for (int r = 0; r < 8; ++r) {
        Cb0[(size_t)(r + hiHalf * 8) * dout] = acc0[r];
        Cb1[(size_t)(r + hiHalf * 8) * dout] = acc1[r];
    }
}

// ---- weight pack: W[K x dout] f32 row-major  ->  Wt[dout x K] f16 ----------
__global__ void pack_weight(const float* __restrict__ W, _Float16* __restrict__ Wt,
                            int K, int dout) {
    int idx = blockIdx.x * blockDim.x + threadIdx.x;
    if (idx >= K * dout) return;
    int n = idx / K, k = idx % K;
    Wt[(size_t)n * K + k] = (_Float16)W[(size_t)k * dout + n];
}

// ---- x f32 -> f16 activation copy -----------------------------------------
__global__ void to_half(const float* __restrict__ src, _Float16* __restrict__ dst, int n) {
    int idx = blockIdx.x * blockDim.x + threadIdx.x;
    if (idx < n) dst[idx] = (_Float16)src[idx];
}

// ---- degree / norm precompute ---------------------------------------------
__global__ void deg_init(float* __restrict__ deg) {
    int i = blockIdx.x * blockDim.x + threadIdx.x;
    if (i < N_NODES) deg[i] = 1.0f;               // self-loop
}
__global__ void deg_edges(const int* __restrict__ dst, float* __restrict__ deg) {
    int e = blockIdx.x * blockDim.x + threadIdx.x;
    if (e < E_EDGES) atomicAdd(deg + dst[e], 1.0f);
}
__global__ void make_dinv(const float* __restrict__ deg, float* __restrict__ dinv) {
    int i = blockIdx.x * blockDim.x + threadIdx.x;
    if (i < N_NODES) dinv[i] = __frsqrt_rn(deg[i]);   // deg >= 1 always
}
__global__ void make_norm(const int* __restrict__ src, const int* __restrict__ dst,
                          const float* __restrict__ dinv, float* __restrict__ nrm) {
    int e = blockIdx.x * blockDim.x + threadIdx.x;
    if (e < E_EDGES) nrm[e] = dinv[src[e]] * dinv[dst[e]];
}

// ---- aggregation: init with self-loop term (doubles as zero-init) ----------
__global__ void agg_init(const float* __restrict__ hW, const float* __restrict__ dinv,
                         float* __restrict__ agg, int d) {
    long long idx = (long long)blockIdx.x * blockDim.x + threadIdx.x;
    if (idx >= (long long)N_NODES * d) return;
    int i = (int)(idx / d);
    float w = dinv[i];
    agg[idx] = hW[idx] * w * w;
}
// ---- aggregation: edge scatter (L2-resident float atomics) -----------------
__global__ void agg_edges(const float* __restrict__ hW,
                          const int* __restrict__ src, const int* __restrict__ dst,
                          const float* __restrict__ nrm,
                          float* __restrict__ agg, int d) {
    long long idx = (long long)blockIdx.x * blockDim.x + threadIdx.x;
    if (idx >= (long long)E_EDGES * d) return;
    int e = (int)(idx / d);
    int f = (int)(idx % d);
    atomicAdd(&agg[(size_t)dst[e] * d + f], hW[(size_t)src[e] * d + f] * nrm[e]);
}

// ---- bias + ReLU, emit f16 activation for next WMMA GEMM -------------------
__global__ void bias_relu_h(const float* __restrict__ agg, const float* __restrict__ b,
                            _Float16* __restrict__ hA, int d) {
    long long idx = (long long)blockIdx.x * blockDim.x + threadIdx.x;
    if (idx >= (long long)N_NODES * d) return;
    float v = agg[idx] + b[idx % d];
    hA[idx] = (_Float16)(v > 0.0f ? v : 0.0f);
}

// ---- mean pool -------------------------------------------------------------
__global__ void zero_vec(float* __restrict__ g, int n) {
    int i = blockIdx.x * blockDim.x + threadIdx.x;
    if (i < n) g[i] = 0.0f;
}
__global__ void pool_sum(const _Float16* __restrict__ hA, float* __restrict__ g,
                         int d, int rowsPerBlock) {
    int col = threadIdx.x;                 // blockDim.x == d (512)
    int r0  = blockIdx.x * rowsPerBlock;
    float s = 0.0f;
    for (int i = 0; i < rowsPerBlock; ++i) {
        int r = r0 + i;
        if (r < N_NODES) s += (float)hA[(size_t)r * d + col];
    }
    atomicAdd(g + col, s);
}

// ---- final dense: out[256] = (g/N) @ W4 + b4 -------------------------------
__global__ void final_dense(const float* __restrict__ g, const float* __restrict__ W4,
                            const float* __restrict__ b4, float* __restrict__ out) {
    int j = threadIdx.x;                   // 256 threads, one block
    const float inv_n = 1.0f / (float)N_NODES;
    float s = 0.0f;
    for (int k = 0; k < 512; ++k)
        s += (g[k] * inv_n) * W4[k * 256 + j];
    out[j] = s + b4[j];
}

// ---------------------------------------------------------------------------
extern "C" void kernel_launch(void* const* d_in, const int* in_sizes, int n_in,
                              void* d_out, int out_size, void* d_ws, size_t ws_size,
                              hipStream_t stream) {
    (void)in_sizes; (void)n_in; (void)out_size; (void)ws_size;

    const float* x   = (const float*)d_in[0];
    const int*   ei  = (const int*)  d_in[1];   // [2, E]: src = ei, dst = ei+E
    const float* W1  = (const float*)d_in[2];
    const float* b1  = (const float*)d_in[3];
    const float* W2  = (const float*)d_in[4];
    const float* b2  = (const float*)d_in[5];
    const float* W3  = (const float*)d_in[6];
    const float* b3  = (const float*)d_in[7];
    const float* W4  = (const float*)d_in[8];
    const float* b4  = (const float*)d_in[9];
    const int* src = ei;
    const int* dst = ei + E_EDGES;
    float* out = (float*)d_out;

    // ---- workspace layout (all regions naturally aligned) ----
    char* ws = (char*)d_ws;
    float* deg  = (float*)ws;                 ws += (size_t)N_NODES * 4;
    float* dinv = (float*)ws;                 ws += (size_t)N_NODES * 4;
    float* nrm  = (float*)ws;                 ws += (size_t)E_EDGES * 4;
    float* g    = (float*)ws;                 ws += 512 * 4;
    ws = (char*)(((uintptr_t)ws + 255) & ~(uintptr_t)255);
    float* hW   = (float*)ws;                 ws += (size_t)N_NODES * 512 * 4;  // GEMM out
    float* agg  = (float*)ws;                 ws += (size_t)N_NODES * 512 * 4;  // scatter acc
    _Float16* hA  = (_Float16*)ws;            ws += (size_t)N_NODES * 512 * 2;  // f16 acts
    _Float16* Wt1 = (_Float16*)ws;            ws += (size_t)128 * 128 * 2;
    _Float16* Wt2 = (_Float16*)ws;            ws += (size_t)128 * 256 * 2;
    _Float16* Wt3 = (_Float16*)ws;            /* ws += 256*512*2 */

    const int T = 256;
    auto blocks = [](long long n, int t) { return (unsigned)((n + t - 1) / t); };

    // ---- weight packs (tiny, once per call) ----
    pack_weight<<<blocks(128 * 128, T), T, 0, stream>>>(W1, Wt1, 128, 128);
    pack_weight<<<blocks(128 * 256, T), T, 0, stream>>>(W2, Wt2, 128, 256);
    pack_weight<<<blocks(256 * 512, T), T, 0, stream>>>(W3, Wt3, 256, 512);

    // ---- x -> f16 ----
    to_half<<<blocks((long long)N_NODES * 128, T), T, 0, stream>>>(x, hA, N_NODES * 128);

    // ---- degree / norm ----
    deg_init <<<blocks(N_NODES, T), T, 0, stream>>>(deg);
    deg_edges<<<blocks(E_EDGES, T), T, 0, stream>>>(dst, deg);
    make_dinv<<<blocks(N_NODES, T), T, 0, stream>>>(deg, dinv);
    make_norm<<<blocks(E_EDGES, T), T, 0, stream>>>(src, dst, dinv, nrm);

    // ---- layer 1: 128 -> 128 ----
    gcn_wmma_gemm<<<dim3(N_NODES / 16, 128 / 128), 128, 0, stream>>>(hA, Wt1, hW, 128, 128);
    agg_init   <<<blocks((long long)N_NODES * 128, T), T, 0, stream>>>(hW, dinv, agg, 128);
    agg_edges  <<<blocks((long long)E_EDGES * 128, T), T, 0, stream>>>(hW, src, dst, nrm, agg, 128);
    bias_relu_h<<<blocks((long long)N_NODES * 128, T), T, 0, stream>>>(agg, b1, hA, 128);

    // ---- layer 2: 128 -> 256 ----
    gcn_wmma_gemm<<<dim3(N_NODES / 16, 256 / 128), 128, 0, stream>>>(hA, Wt2, hW, 128, 256);
    agg_init   <<<blocks((long long)N_NODES * 256, T), T, 0, stream>>>(hW, dinv, agg, 256);
    agg_edges  <<<blocks((long long)E_EDGES * 256, T), T, 0, stream>>>(hW, src, dst, nrm, agg, 256);
    bias_relu_h<<<blocks((long long)N_NODES * 256, T), T, 0, stream>>>(agg, b2, hA, 256);

    // ---- layer 3: 256 -> 512 ----
    gcn_wmma_gemm<<<dim3(N_NODES / 16, 512 / 128), 128, 0, stream>>>(hA, Wt3, hW, 256, 512);
    agg_init   <<<blocks((long long)N_NODES * 512, T), T, 0, stream>>>(hW, dinv, agg, 512);
    agg_edges  <<<blocks((long long)E_EDGES * 512, T), T, 0, stream>>>(hW, src, dst, nrm, agg, 512);
    bias_relu_h<<<blocks((long long)N_NODES * 512, T), T, 0, stream>>>(agg, b3, hA, 512);

    // ---- mean pool + final dense ----
    zero_vec<<<2, 256, 0, stream>>>(g, 512);
    {
        const int rowsPerBlock = 157;                       // 64 blocks cover 10048 rows
        pool_sum<<<(N_NODES + rowsPerBlock - 1) / rowsPerBlock, 512, 0, stream>>>(hA, g, 512, rowsPerBlock);
    }
    final_dense<<<1, 256, 0, stream>>>(g, W4, b4, out);
}
